// EnAttention_18270790877647
// MI455X (gfx1250) — matrix operand order repaired and checked by reading, not compiled, not run on hardware
//
#include <hip/hip_runtime.h>
#include <hip/hip_bf16.h>
#include <stdint.h>

#define N_NODES 50000
#define N_EDGES 800000
#define IN_NF 64
#define OUT_NF 64
#define HID 128
#define NH 4
#define HD 256

typedef _Float16 v16h __attribute__((ext_vector_type(16)));
typedef _Float16 half8 __attribute__((ext_vector_type(8)));
typedef float v8f __attribute__((ext_vector_type(8)));

__device__ __forceinline__ float silu_f(float v) {
  return v * __builtin_amdgcn_rcpf(1.f + __expf(-v));
}
__device__ __forceinline__ float sigm_f(float v) {
  return __builtin_amdgcn_rcpf(1.f + __expf(-v));
}

// ordered-uint encoding so atomicMax(u32) == float max
__device__ __forceinline__ unsigned fkey(float f) {
  unsigned b = __float_as_uint(f);
  return (b & 0x80000000u) ? ~b : (b | 0x80000000u);
}
__device__ __forceinline__ float fdec(unsigned k) {
  unsigned b = (k & 0x80000000u) ? (k & 0x7FFFFFFFu) : ~k;
  return __uint_as_float(b);
}

// ---------------------------------------------------------------------------
// Weight prep: Wt[n*Kpad + k] = (k < Kreal) ? (f16)W[k*Nout + n] : 0
// ---------------------------------------------------------------------------
__global__ void k_prep_w(const float* __restrict__ W, int Kreal, int Nout, int Kpad,
                         _Float16* __restrict__ Wt) {
  int idx = blockIdx.x * blockDim.x + threadIdx.x;
  if (idx >= Nout * Kpad) return;
  int n = idx / Kpad, k = idx % Kpad;
  Wt[idx] = (k < Kreal) ? (_Float16)W[k * Nout + n] : (_Float16)0.f;
}

// ---------------------------------------------------------------------------
// WMMA GEMM:  C[M, NW*16] = act(A[M,KPAD](f16) @ Wt^T + bias)
// Wt is [NW*16][KPAD] f16 (pre-transposed/padded). RT row tiles (16 rows) per
// block; B fragments preloaded into registers once and reused across tiles.
// One wave per 16-col tile. Fully unrolled k-loop -> back-to-back
// v_wmma_f32_16x16x32_f16. f16 outputs are transposed through a wave-private
// LDS tile (DScnt-ordered, no barrier) -> one coalesced b128 store per lane.
// A fragment (16-bit A 16x32): lane<16 holds K 0..7,16..23; lane>=16 holds
// K 8..15,24..31 (contiguous half8 pairs). B fragment: lane l -> N=l%16,
// element e -> K = k0 + 8*(l/16) + e%8 + 16*(e/8) (same as A). C/D: vgpr r -> M = r+8*(l/16), N=l%16.
// ---------------------------------------------------------------------------
template <int KPAD, int NW, int RT, bool SILU, bool OUTF16>
__global__ void wmma_gemm(const _Float16* __restrict__ A,
                          const _Float16* __restrict__ Wt, int M,
                          const float* __restrict__ bias,
                          _Float16* __restrict__ Ch, float* __restrict__ Cf) {
  constexpr int NOUT = NW * 16;
  __shared__ uint4 As4[RT * 16 * KPAD / 8];
  __shared__ uint4 Cs4[NW * 32];  // per-wave 16x16 f16 tile (512B each)
  _Float16* As = (_Float16*)As4;
  const int tid = threadIdx.x;
  const int lane = tid & 31;
  const int wave = tid >> 5;
  const int rowBase = blockIdx.x * (16 * RT);

  // vectorized cooperative stage of the (RT*16) x KPAD A tile into LDS
  const int chunksPerRow = KPAD / 8;  // uint4 chunks per row
  const int totalChunks = RT * 16 * chunksPerRow;
  const int availChunks = (M - rowBase) * chunksPerRow;
  const uint4* src = (const uint4*)(A + rowBase * KPAD);
  for (int i = tid; i < totalChunks; i += blockDim.x) {
    uint4 z = {0u, 0u, 0u, 0u};
    As4[i] = (i < availChunks) ? src[i] : z;
  }
  __syncthreads();

  const int mloc = lane & 15;
  const int khalf = lane >> 4;  // 0 or 1
  const int ncol = wave * 16 + mloc;
  const _Float16* wp = Wt + ncol * KPAD;

  // preload all B fragments; they stay in registers across the RT row tiles
  v16h bfrag[KPAD / 32];
#pragma unroll
  for (int ks = 0; ks < KPAD / 32; ++ks) {
    const _Float16* bp = wp + ks * 32 + khalf * 8;
    ((half8*)&bfrag[ks])[0] = *(const half8*)(bp);
    ((half8*)&bfrag[ks])[1] = *(const half8*)(bp + 16);
  }
  const float bv = bias[ncol];
  _Float16* cs = (_Float16*)Cs4 + wave * 256;

#pragma unroll
  for (int rt = 0; rt < RT; ++rt) {
    const int mt = rowBase + rt * 16;
    if (mt >= M) break;
    v8f acc = {};
#pragma unroll
    for (int ks = 0; ks < KPAD / 32; ++ks) {
      v16h a;
      const _Float16* ap = &As[(rt * 16 + mloc) * KPAD + ks * 32 + khalf * 8];
      ((half8*)&a)[0] = *(const half8*)(ap);       // K = 32ks + 8*khalf + 0..7
      ((half8*)&a)[1] = *(const half8*)(ap + 16);  // K = 32ks + 16 + 8*khalf + 0..7
      acc = __builtin_amdgcn_wmma_f32_16x16x32_f16(false, a, false, bfrag[ks],
                                                   (short)0, acc, false, false);
    }
    if (OUTF16) {
      // transpose through wave-private LDS tile, then coalesced b128 stores
#pragma unroll
      for (int r = 0; r < 8; ++r) {
        float v = acc[r] + bv;
        if (SILU) v = silu_f(v);
        cs[(khalf * 8 + r) * 16 + mloc] = (_Float16)v;  // row-major 16x16 tile
      }
      const int row = lane >> 1, part = lane & 1;  // 2 lanes per row
      uint4 val = *(const uint4*)&cs[row * 16 + part * 8];
      *(uint4*)&Ch[(mt + row) * NOUT + wave * 16 + part * 8] = val;
    } else {
#pragma unroll
      for (int r = 0; r < 8; ++r) {
        int m = mt + r + khalf * 8;
        float v = acc[r] + bv;
        if (SILU) v = silu_f(v);
        Cf[m * NOUT + ncol] = v;
      }
    }
  }
}

// ---------------------------------------------------------------------------
// small helper kernels
// ---------------------------------------------------------------------------
__global__ void k_stats(const float* __restrict__ x, float* __restrict__ stats) {
  int i = blockIdx.x * blockDim.x + threadIdx.x;
  if (i < N_NODES) {
#pragma unroll
    for (int d = 0; d < 3; ++d) {
      float v = x[i * 3 + d];
      atomicAdd(&stats[d], v);
      atomicAdd(&stats[3 + d], v * v);
    }
  }
}

__global__ void k_xn(const float* __restrict__ x, const float* __restrict__ stats,
                     float* __restrict__ xn) {
  int i = blockIdx.x * blockDim.x + threadIdx.x;
  if (i < N_NODES) {
#pragma unroll
    for (int d = 0; d < 3; ++d) {
      float mean = stats[d] / (float)N_NODES;
      float var = (stats[3 + d] - (float)N_NODES * mean * mean) / (float)(N_NODES - 1);
      float sd = sqrtf(fmaxf(var, 0.f));
      xn[i * 3 + d] = (x[i * 3 + d] - mean) / (sd + 1e-8f);
    }
  }
}

__global__ void k_f32_to_f16(const float* __restrict__ src, _Float16* __restrict__ dst,
                             int n) {
  int i = blockIdx.x * blockDim.x + threadIdx.x;
  if (i < n) dst[i] = (_Float16)src[i];
}

__global__ void k_rel(const int* __restrict__ ei, const float* __restrict__ x,
                      float* __restrict__ relp, float* __restrict__ reld) {
  int e = blockIdx.x * blockDim.x + threadIdx.x;
  if (e < N_EDGES) {
    int r = ei[e], c = ei[N_EDGES + e];
    float d2 = 0.f;
#pragma unroll
    for (int d = 0; d < 3; ++d) {
      float v = x[r * 3 + d] - x[c * 3 + d];
      relp[e * 3 + d] = v;
      d2 += v * v;
    }
    reld[e] = d2;
  }
}

// edge_attr = [h[row](64), h[col](64), rel_dist(1), xn[row](3), xn[col](3)] pad->144
__global__ void k_gather_ea(const int* __restrict__ ei, const float* __restrict__ h,
                            const float* __restrict__ reld, const float* __restrict__ xn,
                            _Float16* __restrict__ ea) {
  int idx = blockIdx.x * blockDim.x + threadIdx.x;
  if (idx >= N_EDGES * 144) return;  // 115.2M < 2^31
  int e = idx / 144, c = idx % 144;
  int r = ei[e], cl = ei[N_EDGES + e];
  float v;
  if (c < 64)        v = h[r * 64 + c];
  else if (c < 128)  v = h[cl * 64 + (c - 64)];
  else if (c == 128) v = reld[e];
  else if (c < 132)  v = xn[r * 3 + (c - 129)];
  else if (c < 135)  v = xn[cl * 3 + (c - 132)];
  else               v = 0.f;
  ea[idx] = (_Float16)v;
}

// attn logits: qk/8 + pos_enc(rel_dist) + ewt(ef), mask, segment-max via atomicMax
__global__ void k_logits(const int* __restrict__ ei, const unsigned char* __restrict__ mask,
                         const _Float16* __restrict__ q16, const _Float16* __restrict__ k16,
                         const _Float16* __restrict__ ef, const float* __restrict__ reld,
                         const float* __restrict__ pw1, const float* __restrict__ pb1,
                         const float* __restrict__ pw2, const float* __restrict__ pb2,
                         const float* __restrict__ eww, const float* __restrict__ ewb,
                         float* __restrict__ attn, unsigned* __restrict__ amaxk) {
  int e = blockIdx.x * blockDim.x + threadIdx.x;
  if (e >= N_EDGES) return;
  int r = ei[e], c = ei[N_EDGES + e];
  float acc[NH];
#pragma unroll
  for (int hh = 0; hh < NH; ++hh) acc[hh] = pb2[hh] + ewb[hh];
  // pos_enc: 1 -> 64 (silu) -> 4
  float rd = reld[e];
  for (int j = 0; j < 64; ++j) {
    float t = silu_f(rd * pw1[j] + pb1[j]);
#pragma unroll
    for (int hh = 0; hh < NH; ++hh) acc[hh] += t * pw2[j * NH + hh];
  }
  // ewt: ef(128) @ eww(128x4)
  for (int j = 0; j < HID; ++j) {
    float f = (float)ef[e * HID + j];
#pragma unroll
    for (int hh = 0; hh < NH; ++hh) acc[hh] += f * eww[j * NH + hh];
  }
  bool m = mask[e] != 0;
#pragma unroll
  for (int hh = 0; hh < NH; ++hh) {
    float dot = 0.f;
    for (int d = 0; d < 64; ++d)
      dot += (float)q16[r * HD + hh * 64 + d] * (float)k16[c * HD + hh * 64 + d];
    float a = dot * 0.125f + acc[hh];
    if (!m) a = -1e9f;
    attn[e * NH + hh] = a;
    atomicMax(&amaxk[r * NH + hh], fkey(a));
  }
}

__global__ void k_expsum(const int* __restrict__ ei, const unsigned* __restrict__ amaxk,
                         float* __restrict__ attn, float* __restrict__ asum) {
  int idx = blockIdx.x * blockDim.x + threadIdx.x;
  if (idx >= N_EDGES * NH) return;
  int e = idx >> 2, hh = idx & 3;
  int r = ei[e];
  float a = __expf(attn[idx] - fdec(amaxk[r * NH + hh]));
  attn[idx] = a;
  atomicAdd(&asum[r * NH + hh], a);
}

__global__ void k_norm(const int* __restrict__ ei, const float* __restrict__ asum,
                       float* __restrict__ attn) {
  int idx = blockIdx.x * blockDim.x + threadIdx.x;
  if (idx >= N_EDGES * NH) return;
  int e = idx >> 2, hh = idx & 3;
  attn[idx] = attn[idx] / (asum[ei[e] * NH + hh] + 1e-8f);
}

// msg_in = [ef(128), attn_mean(1), rel_pos(3), xn[row](3), xn[col](3)] pad->144
__global__ void k_gather_mi(const int* __restrict__ ei, const _Float16* __restrict__ ef,
                            const float* __restrict__ attn, const float* __restrict__ relp,
                            const float* __restrict__ xn, _Float16* __restrict__ mi) {
  int idx = blockIdx.x * blockDim.x + threadIdx.x;
  if (idx >= N_EDGES * 144) return;
  int e = idx / 144, c = idx % 144;
  if (c < 128) { mi[idx] = ef[e * HID + c]; return; }
  int r = ei[e], cl = ei[N_EDGES + e];
  float v;
  if (c == 128)
    v = 0.25f * (attn[e * 4] + attn[e * 4 + 1] + attn[e * 4 + 2] + attn[e * 4 + 3]);
  else if (c < 132) v = relp[e * 3 + (c - 129)];
  else if (c < 135) v = xn[r * 3 + (c - 132)];
  else if (c < 138) v = xn[cl * 3 + (c - 135)];
  else              v = 0.f;
  mi[idx] = (_Float16)v;
}

// gates/cwts/cg heads fused with coordinate scatter
__global__ void k_coord(const int* __restrict__ ei, const float* __restrict__ x,
                        const float* __restrict__ relp,
                        const _Float16* __restrict__ msgs, const _Float16* __restrict__ tc,
                        const _Float16* __restrict__ tx,
                        const float* __restrict__ gw, const float* __restrict__ gb,
                        const float* __restrict__ cw2, const float* __restrict__ cb2,
                        const float* __restrict__ xw2, const float* __restrict__ xb2,
                        const float* __restrict__ cwts_w, float* __restrict__ outc) {
  int e = blockIdx.x * blockDim.x + threadIdx.x;
  if (e >= N_EDGES) return;
  float g[NH], cw[NH], cg[NH];
#pragma unroll
  for (int hh = 0; hh < NH; ++hh) { g[hh] = gb[hh]; cw[hh] = cb2[hh]; cg[hh] = xb2[hh]; }
  for (int j = 0; j < HID; ++j) {
    float m = (float)msgs[e * HID + j];
    float a = (float)tc[e * HID + j];
    float b = (float)tx[e * HID + j];
#pragma unroll
    for (int hh = 0; hh < NH; ++hh) {
      g[hh]  += m * gw[j * NH + hh];
      cw[hh] += a * cw2[j * NH + hh];
      cg[hh] += b * xw2[j * NH + hh];
    }
  }
  float s = 0.f, cs = 0.f;
#pragma unroll
  for (int hh = 0; hh < NH; ++hh) {
    s += sigm_f(g[hh]) * cw[hh] * cwts_w[hh];
    cs += cg[hh];
  }
  const float inv = 1.f / (float)(N_NODES - 1);
  int r = ei[e], c = ei[N_EDGES + e];
  int ep = (e == 0) ? (N_EDGES - 1) : (e - 1);
  float a0 = relp[e * 3], a1 = relp[e * 3 + 1], a2 = relp[e * 3 + 2];
  float b0 = relp[ep * 3], b1 = relp[ep * 3 + 1], b2 = relp[ep * 3 + 2];
  float cr[3] = {a1 * b2 - a2 * b1, a2 * b0 - a0 * b2, a0 * b1 - a1 * b0};
#pragma unroll
  for (int d = 0; d < 3; ++d) {
    float base = 0.9f * relp[e * 3 + d] + 0.1f * x[c * 3 + d];
    atomicAdd(&outc[r * 3 + d], (s * base + cs * cr[d]) * inv);
  }
}

// wv[row] += attn * v[col]
__global__ void k_wv(const int* __restrict__ ei, const float* __restrict__ attn,
                     const _Float16* __restrict__ v16, float* __restrict__ wv) {
  int idx = blockIdx.x * blockDim.x + threadIdx.x;
  if (idx >= N_EDGES * 256) return;  // 204.8M < 2^31
  int e = idx >> 8, j = idx & 255;
  int hh = j >> 6;
  int r = ei[e], c = ei[N_EDGES + e];
  float val = attn[e * NH + hh] * (float)v16[c * HD + j];
  atomicAdd(&wv[r * HD + j], val);
}

// ---------------------------------------------------------------------------
extern "C" void kernel_launch(void* const* d_in, const int* in_sizes, int n_in,
                              void* d_out, int out_size, void* d_ws, size_t ws_size,
                              hipStream_t stream) {
  (void)in_sizes; (void)n_in; (void)out_size; (void)ws_size;
  const float* h  = (const float*)d_in[0];
  const float* x  = (const float*)d_in[1];
  const int*   ei = (const int*)d_in[2];
  const unsigned char* mask = (const unsigned char*)d_in[3];
  const float *Wq = (const float*)d_in[4],  *bq = (const float*)d_in[5];
  const float *Wk = (const float*)d_in[6],  *bk = (const float*)d_in[7];
  const float *Wv = (const float*)d_in[8],  *bv = (const float*)d_in[9];
  const float *Wo = (const float*)d_in[10], *bo = (const float*)d_in[11];
  const float *pw1 = (const float*)d_in[12], *pb1 = (const float*)d_in[13];
  const float *pw2 = (const float*)d_in[14], *pb2 = (const float*)d_in[15];
  const float *ew1 = (const float*)d_in[16], *eb1 = (const float*)d_in[17];
  const float *ew2 = (const float*)d_in[18], *eb2 = (const float*)d_in[19];
  const float *eww = (const float*)d_in[20], *ewb = (const float*)d_in[21];
  const float *mw1 = (const float*)d_in[22], *mb1 = (const float*)d_in[23];
  const float *mw2 = (const float*)d_in[24], *mb2 = (const float*)d_in[25];
  const float *gw  = (const float*)d_in[26], *gb  = (const float*)d_in[27];
  const float *cw1 = (const float*)d_in[28], *cb1 = (const float*)d_in[29];
  const float *cw2 = (const float*)d_in[30], *cb2 = (const float*)d_in[31];
  const float *coordw = (const float*)d_in[32];
  const float *xw1 = (const float*)d_in[33], *xb1 = (const float*)d_in[34];
  const float *xw2 = (const float*)d_in[35], *xb2 = (const float*)d_in[36];

  float* out  = (float*)d_out;
  float* outc = out + (size_t)N_NODES * OUT_NF;

  // workspace layout
  char* ws = (char*)d_ws;
  size_t off = 0;
  auto take = [&](size_t bytes) -> char* {
    char* p = ws + off;
    off += (bytes + 255) & ~(size_t)255;
    return p;
  };
  float*    stats = (float*)take(6 * 4);
  float*    xn    = (float*)take((size_t)N_NODES * 3 * 4);
  _Float16* h16   = (_Float16*)take((size_t)N_NODES * 64 * 2);
  _Float16* q16   = (_Float16*)take((size_t)N_NODES * HD * 2);
  _Float16* k16   = (_Float16*)take((size_t)N_NODES * HD * 2);
  _Float16* v16   = (_Float16*)take((size_t)N_NODES * HD * 2);
  float*    relp  = (float*)take((size_t)N_EDGES * 3 * 4);
  float*    reld  = (float*)take((size_t)N_EDGES * 4);
  float*    attn  = (float*)take((size_t)N_EDGES * NH * 4);
  unsigned* amaxk = (unsigned*)take((size_t)N_NODES * NH * 4);
  float*    asum  = (float*)take((size_t)N_NODES * NH * 4);
  float*    wv    = (float*)take((size_t)N_NODES * HD * 4);
  _Float16* wv16  = (_Float16*)take((size_t)N_NODES * HD * 2);
  // transposed/padded f16 weights
  _Float16* wtq  = (_Float16*)take((size_t)HD * 64 * 2);
  _Float16* wtk  = (_Float16*)take((size_t)HD * 64 * 2);
  _Float16* wtv  = (_Float16*)take((size_t)HD * 64 * 2);
  _Float16* wto  = (_Float16*)take((size_t)OUT_NF * HD * 2);
  _Float16* wte1 = (_Float16*)take((size_t)HID * 144 * 2);
  _Float16* wte2 = (_Float16*)take((size_t)HID * HID * 2);
  _Float16* wtm1 = (_Float16*)take((size_t)HID * 144 * 2);
  _Float16* wtm2 = (_Float16*)take((size_t)HID * HID * 2);
  _Float16* wtc1 = (_Float16*)take((size_t)HID * HID * 2);
  _Float16* wtx1 = (_Float16*)take((size_t)HID * HID * 2);
  _Float16* big0 = (_Float16*)take((size_t)N_EDGES * 144 * 2);  // ea / mi / tx
  _Float16* big1 = (_Float16*)take((size_t)N_EDGES * HID * 2);  // t1 / tc
  _Float16* big2 = (_Float16*)take((size_t)N_EDGES * HID * 2);  // ef / msgs

  // zero init
  hipMemsetAsync(stats, 0, 6 * 4, stream);
  hipMemsetAsync(amaxk, 0, (size_t)N_NODES * NH * 4, stream);
  hipMemsetAsync(asum, 0, (size_t)N_NODES * NH * 4, stream);
  hipMemsetAsync(wv, 0, (size_t)N_NODES * HD * 4, stream);
  hipMemsetAsync(outc, 0, (size_t)N_NODES * 3 * 4, stream);

  const int B = 256;
  auto blocks = [](long long n, int b) { return (unsigned)((n + b - 1) / b); };

  // weight prep: transpose + pad + f16
  auto prep = [&](const float* W, int Kreal, int Nout, int Kpad, _Float16* Wt) {
    hipLaunchKernelGGL(k_prep_w, dim3(blocks((long long)Nout * Kpad, B)), dim3(B), 0,
                       stream, W, Kreal, Nout, Kpad, Wt);
  };
  prep(Wq, 64, HD, 64, wtq);
  prep(Wk, 64, HD, 64, wtk);
  prep(Wv, 64, HD, 64, wtv);
  prep(Wo, HD, OUT_NF, HD, wto);
  prep(ew1, 135, HID, 144, wte1);
  prep(ew2, HID, HID, HID, wte2);
  prep(mw1, 138, HID, 144, wtm1);
  prep(mw2, HID, HID, HID, wtm2);
  prep(cw1, HID, HID, HID, wtc1);
  prep(xw1, HID, HID, HID, wtx1);

#define GEMM(KP, NWV, SL, OF, A_, Wt_, M_, b_, Ch_, Cf_)                       \
  hipLaunchKernelGGL((wmma_gemm<KP, NWV, 4, SL, OF>), dim3(((M_) + 63) / 64),  \
                     dim3(32 * (NWV)), 0, stream, A_, Wt_, M_, b_, Ch_, Cf_)

  // node statistics + normalized coordinates
  hipLaunchKernelGGL(k_stats, dim3(blocks(N_NODES, B)), dim3(B), 0, stream, x, stats);
  hipLaunchKernelGGL(k_xn, dim3(blocks(N_NODES, B)), dim3(B), 0, stream, x, stats, xn);

  // QKV projections (WMMA)
  hipLaunchKernelGGL(k_f32_to_f16, dim3(blocks((long long)N_NODES * 64, B)), dim3(B), 0,
                     stream, h, h16, N_NODES * 64);
  GEMM(64, 16, false, true, h16, wtq, N_NODES, bq, q16, nullptr);
  GEMM(64, 16, false, true, h16, wtk, N_NODES, bk, k16, nullptr);
  GEMM(64, 16, false, true, h16, wtv, N_NODES, bv, v16, nullptr);

  // edge geometry + edge MLP (WMMA)
  hipLaunchKernelGGL(k_rel, dim3(blocks(N_EDGES, B)), dim3(B), 0, stream, ei, x, relp, reld);
  hipLaunchKernelGGL(k_gather_ea, dim3(blocks((long long)N_EDGES * 144, B)), dim3(B), 0,
                     stream, ei, h, reld, xn, big0);
  GEMM(144, 8, true, true, big0, wte1, N_EDGES, eb1, big1, nullptr);   // silu
  GEMM(128, 8, false, true, big1, wte2, N_EDGES, eb2, big2, nullptr);  // ef

  // attention logits + segment softmax
  hipLaunchKernelGGL(k_logits, dim3(blocks(N_EDGES, B)), dim3(B), 0, stream,
                     ei, mask, q16, k16, big2, reld, pw1, pb1, pw2, pb2, eww, ewb,
                     attn, amaxk);
  hipLaunchKernelGGL(k_expsum, dim3(blocks((long long)N_EDGES * NH, B)), dim3(B), 0,
                     stream, ei, amaxk, attn, asum);
  hipLaunchKernelGGL(k_norm, dim3(blocks((long long)N_EDGES * NH, B)), dim3(B), 0,
                     stream, ei, asum, attn);

  // message MLP (WMMA)
  hipLaunchKernelGGL(k_gather_mi, dim3(blocks((long long)N_EDGES * 144, B)), dim3(B), 0,
                     stream, ei, big2, attn, relp, xn, big0);
  GEMM(144, 8, true, true, big0, wtm1, N_EDGES, mb1, big1, nullptr);   // silu
  GEMM(128, 8, false, true, big1, wtm2, N_EDGES, mb2, big2, nullptr);  // msgs
  GEMM(128, 8, true, true, big2, wtc1, N_EDGES, cb1, big1, nullptr);   // tc
  GEMM(128, 8, true, true, big2, wtx1, N_EDGES, xb1, big0, nullptr);   // tx

  // coordinate update scatter
  hipLaunchKernelGGL(k_coord, dim3(blocks(N_EDGES, B)), dim3(B), 0, stream,
                     ei, x, relp, big2, big1, big0, gw, gb, cw2, cb2, xw2, xb2,
                     coordw, outc);

  // attn-weighted value aggregation + output projection (WMMA)
  hipLaunchKernelGGL(k_wv, dim3(blocks((long long)N_EDGES * HD, B)), dim3(B), 0, stream,
                     ei, attn, v16, wv);
  hipLaunchKernelGGL(k_f32_to_f16, dim3(blocks((long long)N_NODES * HD, B)), dim3(B), 0,
                     stream, wv, wv16, N_NODES * HD);
  GEMM(256, 4, false, false, wv16, wto, N_NODES, bo, nullptr, out);

#undef GEMM
}
